// MoEPipelineStage_58282706206786
// MI455X (gfx1250) — matrix operand order repaired and checked by reading, not compiled, not run on hardware
//
#include <hip/hip_runtime.h>

#define HDIM 1024
#define IDIM 4096
#define NEXP 8
#define ICHUNK 1024
#define NCHUNK (IDIM / ICHUNK)
#define LSTRIDE 48   // LDS row stride in bf16 elems (96B, keeps 16B-aligned fragment loads)

typedef __attribute__((ext_vector_type(16))) __bf16 v16bf;
typedef __attribute__((ext_vector_type(8)))  __bf16 v8bf;
typedef __attribute__((ext_vector_type(8)))  float  v8f;

static __device__ __forceinline__ v16bf cat16(v8bf lo, v8bf hi) {
  return __builtin_shufflevector(lo, hi, 0, 1, 2, 3, 4, 5, 6, 7, 8, 9, 10, 11, 12, 13, 14, 15);
}

static __device__ __forceinline__ v8bf cvt8(const float4 a, const float4 b) {
  v8bf r;
  r[0] = (__bf16)a.x; r[1] = (__bf16)a.y; r[2] = (__bf16)a.z; r[3] = (__bf16)a.w;
  r[4] = (__bf16)b.x; r[5] = (__bf16)b.y; r[6] = (__bf16)b.z; r[7] = (__bf16)b.w;
  return r;
}

// ---------------------------------------------------------------------------
// 0) zero expert counters
// ---------------------------------------------------------------------------
__global__ void zero_cnt_kernel(int* cnt) {
  if (threadIdx.x < NEXP) cnt[threadIdx.x] = 0;
}

// ---------------------------------------------------------------------------
// 1) router: logits -> softmax -> top2 -> assignment lists
//    combine weight quirk (faithful to reference): weight of the k-th selected
//    expert is probs[k] (probability of expert-index k), NOT probs[idx_k].
// ---------------------------------------------------------------------------
__global__ void __launch_bounds__(256)
router_kernel(const float* __restrict__ x, const float* __restrict__ cw, int T,
              int* __restrict__ cnt, int* __restrict__ tokidx,
              float* __restrict__ tokw, int* __restrict__ list_assign) {
  __shared__ float scw[NEXP * HDIM];
  for (int i = threadIdx.x; i < NEXP * HDIM; i += blockDim.x) scw[i] = cw[i];
  __syncthreads();

  const int wid = threadIdx.x >> 5;
  const int lane = threadIdx.x & 31;
  const int t = blockIdx.x * 8 + wid;
  if (t >= T) return;

  float acc[NEXP];
#pragma unroll
  for (int e = 0; e < NEXP; ++e) acc[e] = 0.f;
  const float* xr = x + (size_t)t * HDIM;
  for (int h = lane; h < HDIM; h += 32) {
    const float xv = xr[h];
#pragma unroll
    for (int e = 0; e < NEXP; ++e)
      acc[e] = __builtin_fmaf(xv, scw[e * HDIM + h], acc[e]);
  }
#pragma unroll
  for (int e = 0; e < NEXP; ++e)
#pragma unroll
    for (int s = 16; s > 0; s >>= 1) acc[e] += __shfl_xor(acc[e], s, 32);

  if (lane == 0) {
    float mx = acc[0];
#pragma unroll
    for (int e = 1; e < NEXP; ++e) mx = fmaxf(mx, acc[e]);
    float p[NEXP], sum = 0.f;
#pragma unroll
    for (int e = 0; e < NEXP; ++e) { p[e] = __expf(acc[e] - mx); sum += p[e]; }
    const float inv = 1.f / sum;
#pragma unroll
    for (int e = 0; e < NEXP; ++e) p[e] *= inv;

    int i0 = 0;
#pragma unroll
    for (int e = 1; e < NEXP; ++e) if (p[e] > p[i0]) i0 = e;
    int i1 = (i0 == 0) ? 1 : 0;
#pragma unroll
    for (int e = 0; e < NEXP; ++e) if (e != i0 && p[e] > p[i1]) i1 = e;

    tokidx[2 * t] = i0; tokidx[2 * t + 1] = i1;
    tokw[2 * t] = p[0]; tokw[2 * t + 1] = p[1];   // faithful quirk

    const int pos0 = atomicAdd(&cnt[i0], 1);
    list_assign[i0 * T + pos0] = 2 * t;
    const int pos1 = atomicAdd(&cnt[i1], 1);
    list_assign[i1 * T + pos1] = 2 * t + 1;
  }
}

// ---------------------------------------------------------------------------
// 2) gate/up GEMM (shared A): act[a][icLocal] = silu(x@gw^T+gb) * (x@uw^T+ub)
//    Block tile: BM=64 x BN=128, K step 32. Wave tile: 32x32 per matrix
//    (2 A-frags x 2 B-frags -> 8 WMMA from 6 fragment loads).
// ---------------------------------------------------------------------------
__global__ void __launch_bounds__(256)
gateup_kernel(const float* __restrict__ x,
              const float* __restrict__ gate_w, const float* __restrict__ gate_b,
              const float* __restrict__ up_w, const float* __restrict__ up_b,
              const int* __restrict__ cnt, const int* __restrict__ list_assign,
              __bf16* __restrict__ act, int T, int ic0) {
  const int e = blockIdx.z;
  const int count = cnt[e];
  const int m0 = blockIdx.y * 64;
  if (m0 >= count) return;
  const int nblk = blockIdx.x * 128;  // column base within chunk
  const int n0g = ic0 + nblk;         // global intermediate column base

  __shared__ __bf16 sA[64 * LSTRIDE];
  __shared__ __bf16 sG[128 * LSTRIDE];
  __shared__ __bf16 sU[128 * LSTRIDE];
  __shared__ int sAssign[64];

  const int tid = threadIdx.x;
  if (tid < 64) {
    const int p = m0 + tid;
    sAssign[tid] = (p < count) ? list_assign[e * T + p] : -1;
  }
  __syncthreads();

  const int wid = tid >> 5, lane = tid & 31;
  const int wm = wid & 1;        // 2 wave rows of 32
  const int wn = wid >> 1;       // 4 wave cols of 32
  const int lrow = lane & 15, lhalf = lane >> 4;
  const int kb = lhalf * 8, krun = lhalf * 16;

  v8f gacc[2][2] = {}, uacc[2][2] = {};

  // load mappings
  const int r = tid >> 2;              // 0..63, A tile row
  const int c8 = (tid & 3) * 8;        // 0/8/16/24
  const int r2 = tid >> 1;             // 0..127, B tile row
  const int c16 = (tid & 1) * 16;      // 0/16
  const size_t wBase = (size_t)e * IDIM * HDIM;
  const int a_r = sAssign[r];
  const int tokr = (a_r >= 0) ? (a_r >> 1) : -1;

  for (int kk = 0; kk < HDIM; kk += 32) {
    // A tile: gathered x rows, fp32 -> bf16 (8 elems/thread)
    {
      v8bf av = {};
      if (tokr >= 0) {
        const float* src = x + (size_t)tokr * HDIM + kk + c8;
        av = cvt8(*(const float4*)(src), *(const float4*)(src + 4));
      }
      *(v8bf*)&sA[r * LSTRIDE + c8] = av;
    }
    // B tiles: gate_w / up_w rows (16 elems/thread each)
    {
      const float* gsrc = gate_w + wBase + (size_t)(n0g + r2) * HDIM + kk + c16;
      *(v8bf*)&sG[r2 * LSTRIDE + c16]     = cvt8(*(const float4*)(gsrc),     *(const float4*)(gsrc + 4));
      *(v8bf*)&sG[r2 * LSTRIDE + c16 + 8] = cvt8(*(const float4*)(gsrc + 8), *(const float4*)(gsrc + 12));
      const float* usrc = up_w + wBase + (size_t)(n0g + r2) * HDIM + kk + c16;
      *(v8bf*)&sU[r2 * LSTRIDE + c16]     = cvt8(*(const float4*)(usrc),     *(const float4*)(usrc + 4));
      *(v8bf*)&sU[r2 * LSTRIDE + c16 + 8] = cvt8(*(const float4*)(usrc + 8), *(const float4*)(usrc + 12));
    }
    __syncthreads();

    // fragment loads first, then a back-to-back WMMA burst
    v16bf af[2], bg[2], bu[2];
#pragma unroll
    for (int mi = 0; mi < 2; ++mi) {
      const int row = wm * 32 + mi * 16 + lrow;
      af[mi] = cat16(*(const v8bf*)&sA[row * LSTRIDE + kb],
                     *(const v8bf*)&sA[row * LSTRIDE + 16 + kb]);
    }
#pragma unroll
    for (int ni = 0; ni < 2; ++ni) {
      const int col = wn * 32 + ni * 16 + lrow;
      bg[ni] = cat16(*(const v8bf*)&sG[col * LSTRIDE + kb],
                     *(const v8bf*)&sG[col * LSTRIDE + 16 + kb]);
      bu[ni] = cat16(*(const v8bf*)&sU[col * LSTRIDE + kb],
                     *(const v8bf*)&sU[col * LSTRIDE + 16 + kb]);
    }
#pragma unroll
    for (int mi = 0; mi < 2; ++mi)
#pragma unroll
      for (int ni = 0; ni < 2; ++ni) {
        gacc[mi][ni] = __builtin_amdgcn_wmma_f32_16x16x32_bf16(
            false, af[mi], false, bg[ni], (short)0, gacc[mi][ni], false, false);
        uacc[mi][ni] = __builtin_amdgcn_wmma_f32_16x16x32_bf16(
            false, af[mi], false, bu[ni], (short)0, uacc[mi][ni], false, false);
      }
    __syncthreads();
  }

  // epilogue: silu(gate+gb) * (up+ub) -> bf16 act
#pragma unroll
  for (int ni = 0; ni < 2; ++ni) {
    const int nl = wn * 32 + ni * 16 + lrow;
    const int nglob = n0g + nl;
    const float gb = gate_b[(size_t)e * IDIM + nglob];
    const float ub = up_b[(size_t)e * IDIM + nglob];
#pragma unroll
    for (int mi = 0; mi < 2; ++mi) {
      const v8f g = gacc[mi][ni];
      const v8f u = uacc[mi][ni];
#pragma unroll
      for (int v = 0; v < 8; ++v) {
        const int ml = wm * 32 + mi * 16 + v + 8 * lhalf;
        if (m0 + ml < count) {
          const int a = sAssign[ml];
          const float gv = g[v] + gb;
          const float s = gv / (1.f + __expf(-gv));
          const float aval = s * (u[v] + ub);
          act[(size_t)a * ICHUNK + (nblk + nl)] = (__bf16)aval;
        }
      }
    }
  }
}

// ---------------------------------------------------------------------------
// 3) down GEMM partial: ybuf[a][h] (+)= act[a][:] @ down_w[e][h][:]^T (chunk)
//    Block tile: BM=64 x BN=128. Wave tile 32x32 (2x2 WMMA from 4 frags).
// ---------------------------------------------------------------------------
__global__ void __launch_bounds__(256)
down_kernel(const __bf16* __restrict__ act, const float* __restrict__ down_w,
            const int* __restrict__ cnt, const int* __restrict__ list_assign,
            float* __restrict__ ybuf, int T, int ic0, int first) {
  const int e = blockIdx.z;
  const int count = cnt[e];
  const int m0 = blockIdx.y * 64;
  if (m0 >= count) return;
  const int n0 = blockIdx.x * 128;   // h base

  __shared__ __bf16 sA[64 * LSTRIDE];
  __shared__ __bf16 sB[128 * LSTRIDE];
  __shared__ int sAssign[64];

  const int tid = threadIdx.x;
  if (tid < 64) {
    const int p = m0 + tid;
    sAssign[tid] = (p < count) ? list_assign[e * T + p] : -1;
  }
  __syncthreads();

  const int wid = tid >> 5, lane = tid & 31;
  const int wm = wid & 1;
  const int wn = wid >> 1;
  const int lrow = lane & 15, lhalf = lane >> 4;
  const int kb = lhalf * 8, krun = lhalf * 16;

  v8f yacc[2][2] = {};

  const int r = tid >> 2;
  const int c8 = (tid & 3) * 8;
  const int r2 = tid >> 1;
  const int c16 = (tid & 1) * 16;
  const int a_r = sAssign[r];
  const size_t dwBase = (size_t)e * HDIM * IDIM;

  for (int kk = 0; kk < ICHUNK; kk += 32) {
    // A tile: act rows (already bf16)
    {
      v8bf av = {};
      if (a_r >= 0) av = *(const v8bf*)(act + (size_t)a_r * ICHUNK + kk + c8);
      *(v8bf*)&sA[r * LSTRIDE + c8] = av;
    }
    // B tile: down_w rows, fp32 -> bf16 (16 elems/thread)
    {
      const float* dsrc = down_w + dwBase + (size_t)(n0 + r2) * IDIM + ic0 + kk + c16;
      *(v8bf*)&sB[r2 * LSTRIDE + c16]     = cvt8(*(const float4*)(dsrc),     *(const float4*)(dsrc + 4));
      *(v8bf*)&sB[r2 * LSTRIDE + c16 + 8] = cvt8(*(const float4*)(dsrc + 8), *(const float4*)(dsrc + 12));
    }
    __syncthreads();

    v16bf af[2], bf[2];
#pragma unroll
    for (int mi = 0; mi < 2; ++mi) {
      const int row = wm * 32 + mi * 16 + lrow;
      af[mi] = cat16(*(const v8bf*)&sA[row * LSTRIDE + kb],
                     *(const v8bf*)&sA[row * LSTRIDE + 16 + kb]);
    }
#pragma unroll
    for (int ni = 0; ni < 2; ++ni) {
      const int col = wn * 32 + ni * 16 + lrow;
      bf[ni] = cat16(*(const v8bf*)&sB[col * LSTRIDE + kb],
                     *(const v8bf*)&sB[col * LSTRIDE + 16 + kb]);
    }
#pragma unroll
    for (int mi = 0; mi < 2; ++mi)
#pragma unroll
      for (int ni = 0; ni < 2; ++ni)
        yacc[mi][ni] = __builtin_amdgcn_wmma_f32_16x16x32_bf16(
            false, af[mi], false, bf[ni], (short)0, yacc[mi][ni], false, false);
    __syncthreads();
  }

#pragma unroll
  for (int ni = 0; ni < 2; ++ni) {
    const int nglob = n0 + wn * 32 + ni * 16 + lrow;
#pragma unroll
    for (int mi = 0; mi < 2; ++mi) {
      const v8f yv = yacc[mi][ni];
#pragma unroll
      for (int v = 0; v < 8; ++v) {
        const int ml = wm * 32 + mi * 16 + v + 8 * lhalf;
        if (m0 + ml < count) {
          const int a = sAssign[ml];
          const size_t yi = (size_t)a * HDIM + nglob;
          ybuf[yi] = first ? yv[v] : (ybuf[yi] + yv[v]);
        }
      }
    }
  }
}

// ---------------------------------------------------------------------------
// 4) combine: out[t] = w0*(y[2t]+db[i0]) + w1*(y[2t+1]+db[i1]); aux scalar = 0
// ---------------------------------------------------------------------------
__global__ void combine_kernel(const float* __restrict__ ybuf, const float* __restrict__ down_b,
                               const int* __restrict__ tokidx, const float* __restrict__ tokw,
                               int T, float* __restrict__ out) {
  const int t = blockIdx.x;
  const int i0 = tokidx[2 * t], i1 = tokidx[2 * t + 1];
  const float w0 = tokw[2 * t], w1 = tokw[2 * t + 1];
  const float* y0 = ybuf + (size_t)(2 * t) * HDIM;
  const float* y1 = ybuf + (size_t)(2 * t + 1) * HDIM;
  const float* b0 = down_b + (size_t)i0 * HDIM;
  const float* b1 = down_b + (size_t)i1 * HDIM;
  float* o = out + (size_t)t * HDIM;
  for (int h = threadIdx.x; h < HDIM; h += blockDim.x)
    o[h] = w0 * (y0[h] + b0[h]) + w1 * (y1[h] + b1[h]);
  if (t == 0 && threadIdx.x == 0) out[(size_t)T * HDIM] = 0.f;  // aux output
}

// ---------------------------------------------------------------------------
extern "C" void kernel_launch(void* const* d_in, const int* in_sizes, int n_in,
                              void* d_out, int out_size, void* d_ws, size_t ws_size,
                              hipStream_t stream) {
  const float* x  = (const float*)d_in[0];
  const float* cw = (const float*)d_in[1];
  const float* gw = (const float*)d_in[2];
  const float* gb = (const float*)d_in[3];
  const float* uw = (const float*)d_in[4];
  const float* ub = (const float*)d_in[5];
  const float* dw = (const float*)d_in[6];
  const float* db = (const float*)d_in[7];
  float* out = (float*)d_out;
  const int T = in_sizes[0] / HDIM;   // 16384 tokens

  char* ws = (char*)d_ws;
  size_t o = 0;
  int* cnt = (int*)(ws + o);                o += 256;
  int* tokidx = (int*)(ws + o);             o += (size_t)2 * T * sizeof(int);
  float* tokw = (float*)(ws + o);           o += (size_t)2 * T * sizeof(float);
  int* list_assign = (int*)(ws + o);        o += (size_t)NEXP * T * sizeof(int);
  o = (o + 255) & ~(size_t)255;
  __bf16* act = (__bf16*)(ws + o);          o += (size_t)2 * T * ICHUNK * sizeof(__bf16);
  float* ybuf = (float*)(ws + o);           o += (size_t)2 * T * HDIM * sizeof(float);
  (void)ws_size; (void)n_in; (void)out_size;

  zero_cnt_kernel<<<1, 32, 0, stream>>>(cnt);
  router_kernel<<<(T + 7) / 8, 256, 0, stream>>>(x, cw, T, cnt, tokidx, tokw, list_assign);

  const int mTiles = (T + 63) / 64;
  for (int ch = 0; ch < NCHUNK; ++ch) {
    const int ic0 = ch * ICHUNK;
    gateup_kernel<<<dim3(ICHUNK / 128, mTiles, NEXP), 256, 0, stream>>>(
        x, gw, gb, uw, ub, cnt, list_assign, act, T, ic0);
    down_kernel<<<dim3(HDIM / 128, mTiles, NEXP), 256, 0, stream>>>(
        act, dw, cnt, list_assign, ybuf, T, ic0, ch == 0 ? 1 : 0);
  }
  combine_kernel<<<T, 256, 0, stream>>>(ybuf, db, tokidx, tokw, T, out);
}
